// Bilinear_39350490366313
// MI455X (gfx1250) — matrix operand order repaired and checked by reading, not compiled, not run on hardware
//
#include <hip/hip_runtime.h>

// ---------------------------------------------------------------------------
// Bilinear: out = (X @ W) @ X^T + b     X:(N,D) f32, W:(D,D) f32, out:(N,N) f32
// Strategy: bf16 hi/lo split-GEMM on V_WMMA_F32_16X16X32_BF16 (3 cross terms),
// f32 accumulate -> ~fp32 accuracy at ~8x the f32-WMMA (16x16x4) MAC rate.
// B panels staged in LDS via GLOBAL_LOAD_ASYNC_TO_LDS_B128 (ASYNCcnt),
// double-buffered; single barrier per K-step:
//   wait_asynccnt(0) -> barrier -> issue stage s+1 -> compute(stage s)
// ---------------------------------------------------------------------------

#define BIL_N 8192
#define BIL_D 1024

typedef __attribute__((ext_vector_type(16))) __bf16 v16bf;
typedef __attribute__((ext_vector_type(8)))  __bf16 v8bf;
typedef __attribute__((ext_vector_type(8)))  float  v8f;

// LDS B-panel layout: 128 rows (= output columns), 32 bf16 payload (64B) per
// row, padded to 80B stride for conflict-free b128 reads.
#define BROW_STRIDE 80

// ---- fp32 -> bf16 hi/lo split helpers -------------------------------------
__device__ __forceinline__ void split_f32(float v, __bf16& h, __bf16& l) {
  h = (__bf16)v;                 // RNE
  l = (__bf16)(v - (float)h);    // residual
}

// ---- async global -> LDS copy: one wave instruction moves 16B per lane,
// tracked once per instruction in the wave's ASYNCcnt ------------------------
__device__ __forceinline__ void async_copy_b128(void* lds_dst,
                                                const void* gsrc) {
  // low 32 bits of the generic shared-space address are the LDS byte offset
  unsigned int laddr = (unsigned int)(unsigned long long)lds_dst;
  asm volatile("global_load_async_to_lds_b128 %0, %1, off"
               :
               : "v"(laddr), "v"(gsrc)
               : "memory");
}

// ---- WMMA fragment loaders (layouts per CDNA5 ISA 7.12.2, wave32) ---------
// A fragment 16x32 bf16 (row-major source, leading dim = ldk):
//   lanes 0-15 : row m0+L,    K chunks [k..k+7]   and [k+16..k+23]
//   lanes 16-31: row m0+L-16, K chunks [k+8..k+15] and [k+24..k+31]
__device__ __forceinline__ v16bf load_a_frag(const __bf16* __restrict__ base,
                                             int ldk, int m, int k, int lane) {
  const __bf16* p =
      base + (size_t)(m + (lane & 15)) * ldk + k + ((lane >> 4) << 3);
  v8bf c0 = *(const v8bf*)(p);       // 16 bytes (global_load_b128)
  v8bf c1 = *(const v8bf*)(p + 16);  // 16 bytes
  v16bf r;
#pragma unroll
  for (int i = 0; i < 8; ++i) { r[i] = c0[i]; r[i + 8] = c1[i]; }
  return r;
}

// B fragment 32x16 bf16 out of the staged LDS panel:
//   lanes 0-15 : col n0+L,    K [k..k+7]    and [k+16..k+23]
//   lanes 16-31: col n0+L-16, K [k+8..k+15] and [k+24..k+31]
__device__ __forceinline__ v16bf load_b_frag_lds(const char* sh, int n,
                                                 int lane) {
  const char* p = sh + (n + (lane & 15)) * BROW_STRIDE + ((lane >> 4) << 4);
  v8bf c0 = *(const v8bf*)(p);        // ds_load_b128
  v8bf c1 = *(const v8bf*)(p + 32);   // ds_load_b128
  v16bf r;
#pragma unroll
  for (int i = 0; i < 8; ++i) { r[i] = c0[i]; r[i + 8] = c1[i]; }
  return r;
}

// ---- conversion kernels ---------------------------------------------------
__global__ __launch_bounds__(256) void split_kernel(
    const float* __restrict__ x, __bf16* __restrict__ hi,
    __bf16* __restrict__ lo, int n) {
  int i = blockIdx.x * blockDim.x + threadIdx.x;
  if (i >= n) return;
  __bf16 h, l;
  split_f32(x[i], h, l);
  hi[i] = h;
  lo[i] = l;
}

// W (K x N, row-major) -> WT (N x K, row-major) split, so the GEMM's B loads
// are contiguous along K for both GEMM stages.
__global__ __launch_bounds__(256) void split_transpose_kernel(
    const float* __restrict__ w, __bf16* __restrict__ hi,
    __bf16* __restrict__ lo, int d) {
  int i = blockIdx.x * blockDim.x + threadIdx.x;
  if (i >= d * d) return;
  int r = i / d;  // k
  int c = i % d;  // n
  __bf16 h, l;
  split_f32(w[i], h, l);
  size_t t = (size_t)c * d + r;
  hi[t] = h;
  lo[t] = l;
}

// ---- split-bf16 GEMM core -------------------------------------------------
// C(M x Ncols) = A(M x K) * Bsrc(Ncols x K)^T  [both operands as hi/lo pairs]
// Block tile 128x128, 8 waves arranged 4(m) x 2(n); wave tile 32x64 = 2x4
// WMMA tiles -> 24x v_wmma_f32_16x16x32_bf16 per wave per 32-wide K step.
// B panels (hi+lo) double-buffered in LDS via async copies (4 wave-level
// async instructions per wave per stage; ASYNCcnt +1 each).
// SPLIT_OUT=true : write C re-split into bf16 hi/lo (stage 1 -> feeds stage 2)
// SPLIT_OUT=false: write f32 C + bias (stage 2 -> d_out)
template <bool SPLIT_OUT>
__global__ __launch_bounds__(256) void gemm_bf16x3(
    const __bf16* __restrict__ Ahi, const __bf16* __restrict__ Alo,
    const __bf16* __restrict__ Bhi, const __bf16* __restrict__ Blo,
    int K, int ldc,
    float* __restrict__ Cf, __bf16* __restrict__ Chi, __bf16* __restrict__ Clo,
    const float* __restrict__ bias) {
  // [buffer][hi/lo][128 rows * 80B]
  __shared__ __align__(16) char Bsh[2][2][128 * BROW_STRIDE];

  const int tid = threadIdx.x;
  const int lane = tid & 31;
  const int wave = tid >> 5;
  const int wm = wave & 3;   // 0..3 -> 32-row slabs
  const int wn = wave >> 2;  // 0..1 -> 64-col slabs
  const int r0 = blockIdx.y * 128 + wm * 32;
  const int cblk = blockIdx.x * 128;
  const int c0 = cblk + wn * 64;

  // Async-copy work split: per panel, 512 x 16B chunks; each lane owns chunk
  // column (tid&3) of rows (tid>>2) and (tid>>2)+64 -> 4 wave-asyncs/stage.
  const int crow = tid >> 2;
  const int ccol = tid & 3;  // 16B chunk within the 64B row payload

  const __bf16* bsrc[2] = {Bhi, Blo};

  auto stage = [&](int kk, int buf) {
#pragma unroll
    for (int p = 0; p < 2; ++p) {
#pragma unroll
      for (int h = 0; h < 2; ++h) {
        const int row = crow + h * 64;
        const __bf16* g = bsrc[p] + (size_t)(cblk + row) * K + kk + ccol * 8;
        async_copy_b128(&Bsh[buf][p][row * BROW_STRIDE + ccol * 16], g);
      }
    }
  };

  const v8f vzero = {};
  v8f acc[2][4];
#pragma unroll
  for (int i = 0; i < 2; ++i)
#pragma unroll
    for (int j = 0; j < 4; ++j) acc[i][j] = vzero;

  const int nsteps = K >> 5;
  stage(0, 0);

  for (int s = 0; s < nsteps; ++s) {
    const int k = s << 5;
    const int buf = s & 1;

    // Stage s was the only batch in flight; its copies landed in LDS.
    asm volatile("s_wait_asynccnt 0x0" ::: "memory");
    // Single barrier: (a) everyone's stage-s data is visible, and (b) every
    // wave has finished the previous compute, so buf^1 may be overwritten.
    __syncthreads();
    if (s + 1 < nsteps) stage((s + 1) << 5, buf ^ 1);  // overlaps compute

    v16bf ah[2], al[2], bh[4], bl[4];
#pragma unroll
    for (int i = 0; i < 2; ++i) {
      ah[i] = load_a_frag(Ahi, K, r0 + i * 16, k, lane);
      al[i] = load_a_frag(Alo, K, r0 + i * 16, k, lane);
    }
#pragma unroll
    for (int j = 0; j < 4; ++j) {
      bh[j] = load_b_frag_lds(&Bsh[buf][0][0], wn * 64 + j * 16, lane);
      bl[j] = load_b_frag_lds(&Bsh[buf][1][0], wn * 64 + j * 16, lane);
    }
#pragma unroll
    for (int i = 0; i < 2; ++i)
#pragma unroll
      for (int j = 0; j < 4; ++j) {
        // hi*hi + hi*lo + lo*hi (lo*lo dropped: below f32 noise floor)
        acc[i][j] = __builtin_amdgcn_wmma_f32_16x16x32_bf16(
            false, ah[i], false, bh[j], (short)0, acc[i][j], false, false);
        acc[i][j] = __builtin_amdgcn_wmma_f32_16x16x32_bf16(
            false, ah[i], false, bl[j], (short)0, acc[i][j], false, false);
        acc[i][j] = __builtin_amdgcn_wmma_f32_16x16x32_bf16(
            false, al[i], false, bh[j], (short)0, acc[i][j], false, false);
      }
  }

  float bv = 0.0f;
  if (!SPLIT_OUT) bv = bias[0];

  // C layout (ISA 7.12.2): VGPR r, lane l -> row = r + 8*(l>=16), col = l%16
#pragma unroll
  for (int i = 0; i < 2; ++i)
#pragma unroll
    for (int j = 0; j < 4; ++j) {
      const int col = c0 + j * 16 + (lane & 15);
      const int rbase = r0 + i * 16 + ((lane >> 4) << 3);
#pragma unroll
      for (int r = 0; r < 8; ++r) {
        const size_t idx = (size_t)(rbase + r) * ldc + col;
        const float v = acc[i][j][r];
        if (SPLIT_OUT) {
          __bf16 h, l;
          split_f32(v, h, l);
          Chi[idx] = h;
          Clo[idx] = l;
        } else {
          Cf[idx] = v + bv;
        }
      }
    }
}

// ---------------------------------------------------------------------------
extern "C" void kernel_launch(void* const* d_in, const int* in_sizes, int n_in,
                              void* d_out, int out_size, void* d_ws,
                              size_t ws_size, hipStream_t stream) {
  const float* X = (const float*)d_in[0];  // (N, D)
  const float* W = (const float*)d_in[1];  // (D, D)
  const float* b = (const float*)d_in[2];  // (1,)
  float* out = (float*)d_out;              // (N, N)

  const int N = BIL_N, D = BIL_D;
  char* ws = (char*)d_ws;
  const size_t szX = (size_t)N * D * sizeof(__bf16);  // 16 MB
  const size_t szW = (size_t)D * D * sizeof(__bf16);  // 2 MB
  __bf16* Xhi  = (__bf16*)(ws);
  __bf16* Xlo  = (__bf16*)(ws + szX);
  __bf16* XWhi = (__bf16*)(ws + 2 * szX);
  __bf16* XWlo = (__bf16*)(ws + 3 * szX);
  __bf16* WThi = (__bf16*)(ws + 4 * szX);
  __bf16* WTlo = (__bf16*)(ws + 4 * szX + szW);

  // 1) split X into bf16 hi/lo
  {
    int total = N * D;
    split_kernel<<<(total + 255) / 256, 256, 0, stream>>>(X, Xhi, Xlo, total);
  }
  // 2) W -> W^T, split
  {
    int total = D * D;
    split_transpose_kernel<<<(total + 255) / 256, 256, 0, stream>>>(W, WThi,
                                                                    WTlo, D);
  }
  // 3) XW = X @ W  == Xsplit @ (WT)^T ; re-split to bf16 hi/lo
  gemm_bf16x3<true><<<dim3(D / 128, N / 128), dim3(256), 0, stream>>>(
      Xhi, Xlo, WThi, WTlo, /*K=*/D, /*ldc=*/D, nullptr, XWhi, XWlo, nullptr);
  // 4) out = XW @ X^T + b
  gemm_bf16x3<false><<<dim3(N / 128, N / 128), dim3(256), 0, stream>>>(
      XWhi, XWlo, Xhi, Xlo, /*K=*/D, /*ldc=*/N, out, nullptr, nullptr, b);
}
